// SparseAttention_83236466196835
// MI455X (gfx1250) — matrix operand order repaired and checked by reading, not compiled, not run on hardware
//
#include <hip/hip_runtime.h>

// ---------------------------------------------------------------------------
// Sparse top-k attention for MI455X (gfx1250), wave32 + WMMA bf16 +
// async global->LDS staging (ASYNCcnt path).
// B=4, N=2048, C=512, H=8, DH=64, TOPK=1024.
// ---------------------------------------------------------------------------

#define BATCH 4
#define NTOK  2048
#define CDIM  512
#define HEADS 8
#define DHEAD 64
#define TOPK  1024
#define SCALE 0.125f   // DHEAD^-0.5

typedef __attribute__((ext_vector_type(16))) __bf16        v16bf;
typedef __attribute__((ext_vector_type(8)))  float         v8f;
typedef __attribute__((ext_vector_type(8)))  unsigned int  v8u;
typedef __attribute__((ext_vector_type(4)))  unsigned int  v4u;

// ----------------------------- helpers -------------------------------------

__device__ __forceinline__ unsigned short f32_to_bf16u(float f) {
    unsigned int x = __builtin_bit_cast(unsigned int, f);
    unsigned int r = x + 0x7FFFu + ((x >> 16) & 1u);   // RNE
    return (unsigned short)(r >> 16);
}

// order-preserving f32 <-> u32 mapping (for radix select of k-th largest)
__device__ __forceinline__ unsigned int f32_ord(float f) {
    unsigned int x = __builtin_bit_cast(unsigned int, f);
    return x ^ ((x & 0x80000000u) ? 0xFFFFFFFFu : 0x80000000u);
}
__device__ __forceinline__ float ord_to_f32(unsigned int u) {
    unsigned int x = (u & 0x80000000u) ? (u ^ 0x80000000u) : ~u;
    return __builtin_bit_cast(float, x);
}

__device__ __forceinline__ v16bf combine8(v4u a, v4u b) {
    v8u r;
    r[0] = a[0]; r[1] = a[1]; r[2] = a[2]; r[3] = a[3];
    r[4] = b[0]; r[5] = b[1]; r[6] = b[2]; r[7] = b[3];
    return __builtin_bit_cast(v16bf, r);
}

// A fragment, 16x32 bf16 (MxK), source row-major [M,K] (global or LDS).
// ISA 7.12.2: lanes 0-15 hold M=lane, K = {k0..k0+7, k0+16..k0+23};
//             lanes 16-31 hold M=lane-16, K = {k0+8..k0+15, k0+24..k0+31}.
__device__ __forceinline__ v16bf load_frag_a(const unsigned short* A, int ld,
                                             int m0, int k0, int lane) {
    int m    = m0 + (lane & 15);
    int koff = k0 + ((lane >> 4) & 1) * 8;
    const unsigned short* p = A + (size_t)m * ld + koff;
    v4u lo = *reinterpret_cast<const v4u*>(p);
    v4u hi = *reinterpret_cast<const v4u*>(p + 16);
    return combine8(lo, hi);
}

// B fragment, 32x16 bf16 (KxN), source row-major [N,K] (i.e. B^T contiguous).
// lanes 0-15: N=lane, K={k0..k0+7, k0+16..k0+23}; lanes 16-31: N=lane-16, K={k0+8..k0+15, k0+24..k0+31}.
__device__ __forceinline__ v16bf load_frag_b(const unsigned short* Bp, int ld,
                                             int n0, int k0, int lane) {
    int n  = n0 + (lane & 15);
    int kb = k0 + ((lane >> 4) & 1) * 8;
    const unsigned short* p = Bp + (size_t)n * ld + kb;
    v4u lo = *reinterpret_cast<const v4u*>(p);
    v4u hi = *reinterpret_cast<const v4u*>(p + 16);
    return combine8(lo, hi);
}

__device__ __forceinline__ v8f wmma_bf16(v16bf a, v16bf b, v8f c) {
    return __builtin_amdgcn_wmma_f32_16x16x32_bf16(
        false, a, false, b, (short)0, c, false, false);
}

// Async global->LDS copy of 16 bytes per lane (CDNA5 ASYNCcnt path).
__device__ __forceinline__ void async_g2l_b128(unsigned lds_off,
                                               unsigned long long gaddr) {
    asm volatile("global_load_async_to_lds_b128 %0, %1, off"
                 :: "v"(lds_off), "v"(gaddr) : "memory");
}
__device__ __forceinline__ void wait_asynccnt0() {
    asm volatile("s_wait_asynccnt 0x0" ::: "memory");
}

// ----------------------------- kernels -------------------------------------

__global__ void cvt_f32_bf16_kernel(const float* __restrict__ src,
                                    unsigned short* __restrict__ dst, int n4) {
    int i = blockIdx.x * blockDim.x + threadIdx.x;   // quad index
    if (i < n4) {
        const float* s = src + (size_t)i * 4;
        unsigned long long pk =
              (unsigned long long)f32_to_bf16u(s[0])
            | ((unsigned long long)f32_to_bf16u(s[1]) << 16)
            | ((unsigned long long)f32_to_bf16u(s[2]) << 32)
            | ((unsigned long long)f32_to_bf16u(s[3]) << 48);
        *(unsigned long long*)(dst + (size_t)i * 4) = pk;
    }
}

// Stage a 16-row x 512-col bf16 A slab (16 KB) into LDS with async copies.
// 256 threads * 4 chunks * 16 B = 16384 B.
__device__ __forceinline__ void stage_a_slab(const unsigned short* gA, int m0,
                                             unsigned short* Ash, int tid) {
    const char* gbase = (const char*)(gA + (size_t)m0 * CDIM);
    unsigned lbase = (unsigned)(uintptr_t)Ash;
#pragma unroll
    for (int c = 0; c < 4; ++c) {
        int off = (tid + c * 256) * 16;
        async_g2l_b128(lbase + (unsigned)off,
                       (unsigned long long)(uintptr_t)(gbase + off));
    }
    wait_asynccnt0();
    __syncthreads();
}

// QKV GEMM: x_bf16 [8192,512] @ w_qkv_bf16^T -> scatter into
//   q [B,H,N,DH], k [B,H,N,DH], vT [B,H,DH,N]   (all bf16)
// grid (512, 12), block 256 (8 waves); wave computes one 16x16 C tile.
// A slab shared across waves via async LDS staging.
__global__ void qkv_gemm_kernel(const unsigned short* __restrict__ xb,
                                const unsigned short* __restrict__ wb,
                                unsigned short* __restrict__ qb,
                                unsigned short* __restrict__ kb,
                                unsigned short* __restrict__ vTb) {
    __shared__ __align__(16) unsigned short Ash[16 * CDIM];
    int tid = threadIdx.x;
    int wid = tid >> 5, lane = tid & 31, hi = (lane >> 4) & 1;
    int m0 = blockIdx.x * 16;
    int n0 = (blockIdx.y * 8 + wid) * 16;

    stage_a_slab(xb, m0, Ash, tid);

    v8f c = {0.f, 0.f, 0.f, 0.f, 0.f, 0.f, 0.f, 0.f};
#pragma unroll
    for (int kk = 0; kk < CDIM; kk += 32) {
        v16bf a = load_frag_a(Ash, CDIM, 0, kk, lane);
        v16bf b = load_frag_b(wb, CDIM, n0, kk, lane);
        c = wmma_bf16(a, b, c);
    }

    int col   = n0 + (lane & 15);
    int which = col >> 9;          // 0=q 1=k 2=v
    int rem   = col & 511;
    int h     = rem >> 6;
    int dh    = rem & 63;
#pragma unroll
    for (int j = 0; j < 8; ++j) {
        int row = m0 + hi * 8 + j;
        int bb  = row >> 11;
        int n   = row & (NTOK - 1);
        unsigned short val = f32_to_bf16u(c[j]);
        size_t nd = (size_t)((bb * HEADS + h) * NTOK + n) * DHEAD + dh;
        if (which == 0)       qb[nd] = val;
        else if (which == 1)  kb[nd] = val;
        else vTb[(size_t)((bb * HEADS + h) * DHEAD + dh) * NTOK + n] = val;
    }
}

// Attention: one block per (qtile, head, batch); 8 waves.
// Scores 16x2048 in registers; exact top-K via 4-pass radix select with
// wave-parallel bin scan; masked softmax; split-K P@V on all 8 waves.
__global__ void sparse_attn_kernel(const unsigned short* __restrict__ q,
                                   const unsigned short* __restrict__ k,
                                   const unsigned short* __restrict__ vT,
                                   unsigned short* __restrict__ attn_out) {
    extern __shared__ char smem_raw[];
    unsigned short* Psh    = (unsigned short*)smem_raw;               // 64 KB
    unsigned int*   hist   = (unsigned int*)(smem_raw + 65536);       // 16 KB
    float*          rowmax = (float*)(smem_raw + 65536 + 16384);      // 16
    float*          rowsum = rowmax + 16;                             // 16
    unsigned int*   rowpre = (unsigned int*)(rowsum + 16);            // 16
    int*            rowk   = (int*)(rowpre + 16);                     // 16
    float*          wpart  = (float*)(rowk + 16);                     // 8*16
    float*          wsum   = wpart + 8 * 16;                          // 8*16

    const int tid  = threadIdx.x;
    const int wid  = tid >> 5;
    const int lane = tid & 31;
    const int hi   = (lane >> 4) & 1;

    const int bh = (int)(blockIdx.z * HEADS + blockIdx.y);
    const unsigned short* qh = q  + (size_t)bh * NTOK * DHEAD;  // [tok][dh]
    const unsigned short* kh = k  + (size_t)bh * NTOK * DHEAD;  // [tok][dh]
    const unsigned short* vh = vT + (size_t)bh * DHEAD * NTOK;  // [dh][tok]
    const int m0 = blockIdx.x * 16;

    // --- Phase 1: scores tile S[16,2048] in registers (scaled) -------------
    v16bf qa0 = load_frag_a(qh, DHEAD, m0, 0, lane);
    v16bf qa1 = load_frag_a(qh, DHEAD, m0, 32, lane);

    v8f acc[16];
#pragma unroll
    for (int t = 0; t < 16; ++t) {
        int n0 = (wid + t * 8) * 16;
        v16bf kb0 = load_frag_b(kh, DHEAD, n0, 0, lane);
        v16bf kb1 = load_frag_b(kh, DHEAD, n0, 32, lane);
        v8f c = {0.f, 0.f, 0.f, 0.f, 0.f, 0.f, 0.f, 0.f};
        c = wmma_bf16(qa0, kb0, c);
        c = wmma_bf16(qa1, kb1, c);
#pragma unroll
        for (int j = 0; j < 8; ++j) c[j] *= SCALE;
        acc[t] = c;
    }

    // --- init LDS state + per-wave row maxima -----------------------------
    for (int i2 = tid; i2 < 16 * 256; i2 += 256) hist[i2] = 0u;
    if (tid < 16) { rowpre[tid] = 0u; rowk[tid] = TOPK; rowsum[tid] = 0.f; }

    float lmax[8];
#pragma unroll
    for (int j = 0; j < 8; ++j) lmax[j] = -3.4e38f;
#pragma unroll
    for (int t = 0; t < 16; ++t)
#pragma unroll
        for (int j = 0; j < 8; ++j) lmax[j] = fmaxf(lmax[j], acc[t][j]);
#pragma unroll
    for (int m = 1; m < 16; m <<= 1)
#pragma unroll
        for (int j = 0; j < 8; ++j)
            lmax[j] = fmaxf(lmax[j], __shfl_xor(lmax[j], m, 32));
    if ((lane & 15) == 0)
#pragma unroll
        for (int j = 0; j < 8; ++j) wpart[wid * 16 + hi * 8 + j] = lmax[j];
    __syncthreads();
    if (tid < 16) {
        float mx = wpart[tid];
#pragma unroll
        for (int w = 1; w < 8; ++w) mx = fmaxf(mx, wpart[w * 16 + tid]);
        rowmax[tid] = mx;
    }
    __syncthreads();

    // --- Phase 2: exact K-th largest per row via 4x8-bit radix select -----
    for (int pass = 0; pass < 4; ++pass) {
        int shift = 24 - pass * 8;
        unsigned pr[8];
#pragma unroll
        for (int j = 0; j < 8; ++j) pr[j] = rowpre[hi * 8 + j];
#pragma unroll
        for (int t = 0; t < 16; ++t) {
#pragma unroll
            for (int j = 0; j < 8; ++j) {
                unsigned u = f32_ord(acc[t][j]);
                bool ok = (pass == 0) ||
                          (((u ^ pr[j]) >> (shift + 8)) == 0u);
                if (ok)
                    atomicAdd(&hist[(hi * 8 + j) * 256 + ((u >> shift) & 255u)], 1u);
            }
        }
        __syncthreads();

        // Wave-parallel descending-bin scan: wave w owns rows {w, w+8}.
        // Lane covers 8 bins in descending order; suffix prefix via shfl_up.
        for (int rr = wid; rr < 16; rr += 8) {
            int kl = rowk[rr];
            unsigned cnt[8];
            int csum = 0;
#pragma unroll
            for (int e = 0; e < 8; ++e) {
                cnt[e] = hist[rr * 256 + (255 - (lane * 8 + e))];
                csum += (int)cnt[e];
            }
            int incl = csum;
#pragma unroll
            for (int d2 = 1; d2 < 32; d2 <<= 1) {
                int t2 = __shfl_up(incl, d2, 32);
                if (lane >= d2) incl += t2;
            }
            int excl = incl - csum;
            bool has = (excl < kl) && (excl + csum >= kl);
            if (has) {   // exactly one lane
                int cum = excl;
#pragma unroll
                for (int e = 0; e < 8; ++e) {
                    if (cum + (int)cnt[e] >= kl) {
                        rowk[rr] = kl - cum;
                        rowpre[rr] |= (unsigned)(255 - (lane * 8 + e)) << shift;
                        break;
                    }
                    cum += (int)cnt[e];
                }
            }
        }
        __syncthreads();
        if (pass < 3) {
            for (int i2 = tid; i2 < 16 * 256; i2 += 256) hist[i2] = 0u;
            __syncthreads();
        }
    }

    // --- Phase 3: masked softmax, P -> LDS as bf16 ------------------------
    float mxr[8], thr[8], lsum[8];
#pragma unroll
    for (int j = 0; j < 8; ++j) {
        mxr[j]  = rowmax[hi * 8 + j];
        thr[j]  = ord_to_f32(rowpre[hi * 8 + j]);
        lsum[j] = 0.f;
    }
#pragma unroll
    for (int t = 0; t < 16; ++t)
#pragma unroll
        for (int j = 0; j < 8; ++j) {
            float s = acc[t][j];
            float p = (s >= thr[j]) ? __expf(s - mxr[j]) : 0.f;
            acc[t][j] = p;
            lsum[j] += p;
        }
#pragma unroll
    for (int m = 1; m < 16; m <<= 1)
#pragma unroll
        for (int j = 0; j < 8; ++j) lsum[j] += __shfl_xor(lsum[j], m, 32);
    if ((lane & 15) == 0)
#pragma unroll
        for (int j = 0; j < 8; ++j) wsum[wid * 16 + hi * 8 + j] = lsum[j];
    __syncthreads();
    if (tid < 16) {
        float s2 = 0.f;
#pragma unroll
        for (int w = 0; w < 8; ++w) s2 += wsum[w * 16 + tid];
        rowsum[tid] = s2;
    }
    __syncthreads();

    float inv[8];
#pragma unroll
    for (int j = 0; j < 8; ++j) inv[j] = 1.0f / rowsum[hi * 8 + j];
#pragma unroll
    for (int t = 0; t < 16; ++t) {
        int col = (wid + t * 8) * 16 + (lane & 15);
#pragma unroll
        for (int j = 0; j < 8; ++j)
            Psh[(hi * 8 + j) * NTOK + col] = f32_to_bf16u(acc[t][j] * inv[j]);
    }
    __syncthreads();

    // --- Phase 4: O = P @ V, split-K across all 8 waves -------------------
    // wave = (khalf, nt): nt = dh-tile 0..3, khalf = K half 0..1.
    {
        int nt    = wid & 3;
        int khalf = wid >> 2;
        int n0    = nt * 16;
        v8f o = {0.f, 0.f, 0.f, 0.f, 0.f, 0.f, 0.f, 0.f};
        int kbeg = khalf * (NTOK / 2);
        for (int kk = kbeg; kk < kbeg + NTOK / 2; kk += 32) {
            v16bf pa = load_frag_a(Psh, NTOK, 0, kk, lane);
            v16bf vb = load_frag_b(vh, NTOK, n0, kk, lane);
            o = wmma_bf16(pa, vb, o);
        }
        float* opart = (float*)hist;  // reuse dead histogram region (4 KB)
        if (khalf == 1) {
#pragma unroll
            for (int j = 0; j < 8; ++j)
                opart[nt * 256 + (hi * 8 + j) * 16 + (lane & 15)] = o[j];
        }
        __syncthreads();
        if (khalf == 0) {
            int colc = (int)blockIdx.y * DHEAD + n0 + (lane & 15);
#pragma unroll
            for (int j = 0; j < 8; ++j) {
                float v2 = o[j] + opart[nt * 256 + (hi * 8 + j) * 16 + (lane & 15)];
                size_t row = (size_t)blockIdx.z * NTOK + m0 + hi * 8 + j;
                attn_out[row * CDIM + colc] = f32_to_bf16u(v2);
            }
        }
    }
}

// Output projection: attn_out_bf16 [8192,512] @ w_proj_bf16^T + bias -> f32
// grid (512, 4), block 256; A slab staged in LDS via async copies.
__global__ void proj_gemm_kernel(const unsigned short* __restrict__ A,
                                 const unsigned short* __restrict__ Bw,
                                 const float* __restrict__ bias,
                                 float* __restrict__ out) {
    __shared__ __align__(16) unsigned short Ash[16 * CDIM];
    int tid = threadIdx.x;
    int wid = tid >> 5, lane = tid & 31, hi = (lane >> 4) & 1;
    int m0 = blockIdx.x * 16;
    int n0 = (blockIdx.y * 8 + wid) * 16;

    stage_a_slab(A, m0, Ash, tid);

    v8f c = {0.f, 0.f, 0.f, 0.f, 0.f, 0.f, 0.f, 0.f};
#pragma unroll
    for (int kk = 0; kk < CDIM; kk += 32) {
        v16bf a = load_frag_a(Ash, CDIM, 0, kk, lane);
        v16bf b = load_frag_b(Bw, CDIM, n0, kk, lane);
        c = wmma_bf16(a, b, c);
    }

    int col  = n0 + (lane & 15);
    float bv = bias[col];
#pragma unroll
    for (int j = 0; j < 8; ++j)
        out[(size_t)(m0 + hi * 8 + j) * CDIM + col] = c[j] + bv;
}

// ----------------------------- launch --------------------------------------

extern "C" void kernel_launch(void* const* d_in, const int* in_sizes, int n_in,
                              void* d_out, int out_size, void* d_ws, size_t ws_size,
                              hipStream_t stream) {
    (void)in_sizes; (void)n_in; (void)out_size; (void)ws_size;

    const float* x      = (const float*)d_in[0];   // [4,2048,512]
    const float* w_qkv  = (const float*)d_in[1];   // [1536,512]
    const float* w_proj = (const float*)d_in[2];   // [512,512]
    const float* b_proj = (const float*)d_in[3];   // [512]
    float* out = (float*)d_out;                    // [4,2048,512] f32

    const size_t MTOT = (size_t)BATCH * NTOK;           // 8192
    char* ws = (char*)d_ws;
    size_t off = 0;
    auto take = [&](size_t bytes) {
        char* p = ws + off;
        off += (bytes + 255) & ~(size_t)255;
        return p;
    };
    unsigned short* xb     = (unsigned short*)take(MTOT * CDIM * 2);           // 8 MB
    unsigned short* wqkvb  = (unsigned short*)take((size_t)3 * CDIM * CDIM * 2);
    unsigned short* wprojb = (unsigned short*)take((size_t)CDIM * CDIM * 2);
    unsigned short* qb     = (unsigned short*)take(MTOT * CDIM * 2);           // [B,H,N,DH]
    unsigned short* kb     = (unsigned short*)take(MTOT * CDIM * 2);
    unsigned short* vTb    = (unsigned short*)take(MTOT * CDIM * 2);           // [B,H,DH,N]
    unsigned short* aob    = (unsigned short*)take(MTOT * CDIM * 2);           // [8192,512]

    // 1) f32 -> bf16 conversions (4 elems / thread)
    {
        int n1 = (int)(MTOT * CDIM) / 4;
        cvt_f32_bf16_kernel<<<(n1 + 255) / 256, 256, 0, stream>>>(x, xb, n1);
        int n2 = 3 * CDIM * CDIM / 4;
        cvt_f32_bf16_kernel<<<(n2 + 255) / 256, 256, 0, stream>>>(w_qkv, wqkvb, n2);
        int n3 = CDIM * CDIM / 4;
        cvt_f32_bf16_kernel<<<(n3 + 255) / 256, 256, 0, stream>>>(w_proj, wprojb, n3);
    }

    // 2) QKV projection (M=8192, N=1536, K=512)
    qkv_gemm_kernel<<<dim3(MTOT / 16, 12), 256, 0, stream>>>(xb, wqkvb, qb, kb, vTb);

    // 3) Sparse attention: grid (qtile, head, batch); 83200 B dynamic LDS
    sparse_attn_kernel<<<dim3(NTOK / 16, HEADS, BATCH), 256, 83200, stream>>>(
        qb, kb, vTb, aob);

    // 4) Output projection + bias (M=8192, N=512, K=512)
    proj_gemm_kernel<<<dim3(MTOT / 16, 4), 256, 0, stream>>>(aob, wprojb, b_proj, out);
}
